// Geo_ALIF_SNN_30906584662188
// MI455X (gfx1250) — matrix operand order repaired and checked by reading, not compiled, not run on hardware
//
#include <hip/hip_runtime.h>

// ---------------------------------------------------------------------------
// Geo-ALIF SNN for MI455X (gfx1250), wave32 + WMMA f16->f32 + async LDS loads.
// B=4096, T=320, EEG=64, TH=32, H=16, GEO=18, OUT=4.
// Workspace: [enc f16 (eeg,geo interleaved) 83.89MB][dp_feat f32 2.62MB]
// ---------------------------------------------------------------------------

typedef _Float16 half_t;
typedef __attribute__((ext_vector_type(16))) _Float16 v16h;
typedef __attribute__((ext_vector_type(8)))  _Float16 v8h;
typedef __attribute__((ext_vector_type(8)))  float    v8f;

#define TLEN 320
#define NB   4096

static __device__ __forceinline__ v8f wmma16(v16h a, v16h b, v8f c) {
  // D(16x16 f32) = A(16x32 f16) x B(32x16 f16) + C
  return __builtin_amdgcn_wmma_f32_16x16x32_f16(false, a, false, b, (short)0, c,
                                                false, false);
}

// A-operand gather (16-bit A 16x32): lane holds row M=lane&15;
// lo lanes carry K = 0..15, hi lanes K = 16..31.  Caller passes
// p already offset by (hi?16:0); elements 8..15 live 8 halfs further.
static __device__ __forceinline__ v16h load_a(const half_t* p) {
  v8h lo = *(const v8h*)(p);
  v8h hh = *(const v8h*)(p + 8);
  return __builtin_shufflevector(lo, hh, 0,1,2,3,4,5,6,7,8,9,10,11,12,13,14,15);
}

// ---------------------------------------------------------------------------
// Kernel 1: conv1(5)+BN1+ReLU -> conv2(3)+BN2+ReLU, geo sigmoid gate; the two
// 16-wide results are interleaved per channel into enc[(b*T+t)*32 + 2c + {e,g}]
// One block per batch element; 8 waves.  LDS < 64KB via T-chunked x staging.
// ---------------------------------------------------------------------------

// static LDS byte offsets (all 128B aligned)
#define OFF_H1   0          // (320+2) rows x 32 halfs  = 20608 B  (conv1 out, +guards)
#define OFF_W1   20608      // 5*32*64 halfs            = 20480 B  (phase A)
#define OFF_W2   20608      // 3*16*32 halfs            =  3072 B  (phase B)
#define OFF_WG   23680      // 16*32 halfs              =  1024 B  (phase B)
#define OFF_XG   24704      // 320*32 halfs             = 20480 B  (phase B)
#define OFF_XE   45184      // 132 rows x 64 halfs      = 16896 B  (chunked x_eeg)
#define OFF_CST  62080      // (32+32+16+16+16) floats  =   448 B
#define SMEM_SZ  62528

__global__ __launch_bounds__(256) void k_encode(
    const float* __restrict__ x_eeg, const float* __restrict__ x_geo,
    const float* __restrict__ c1w, const float* __restrict__ c1b,
    const float* __restrict__ bn1s, const float* __restrict__ bn1b,
    const float* __restrict__ bn1m, const float* __restrict__ bn1v,
    const float* __restrict__ c2w, const float* __restrict__ c2b,
    const float* __restrict__ bn2s, const float* __restrict__ bn2b,
    const float* __restrict__ bn2m, const float* __restrict__ bn2v,
    const float* __restrict__ gw, const float* __restrict__ gb,
    half_t* __restrict__ enc_out) {
  __shared__ __align__(128) unsigned char smem[SMEM_SZ];
  half_t* h1  = (half_t*)(smem + OFF_H1);
  half_t* w1l = (half_t*)(smem + OFF_W1);
  half_t* w2l = (half_t*)(smem + OFF_W2);
  half_t* wgl = (half_t*)(smem + OFF_WG);
  half_t* xg  = (half_t*)(smem + OFF_XG);
  half_t* xec = (half_t*)(smem + OFF_XE);
  float* inv1 = (float*)(smem + OFF_CST);
  float* add1 = inv1 + 32;
  float* inv2 = add1 + 32;
  float* add2 = inv2 + 16;
  float* gbia = add2 + 16;

  const int tid  = threadIdx.x;
  const int lane = tid & 31;
  const int wid  = tid >> 5;
  const int n16  = lane & 15;
  const int hi   = lane >> 4;         // 0: lanes 0-15, 1: lanes 16-31
  const int b    = blockIdx.x;

  // ---- phase 0: weights / BN constants / guard rows ----
  for (int idx = tid; idx < 5 * 32 * 64; idx += 256) {
    int k = idx >> 11, o = (idx >> 6) & 31, i = idx & 63;
    w1l[idx] = (half_t)c1w[(o * 64 + i) * 5 + k];     // [tap][o][i]
  }
  if (tid < 32) {
    float iv = bn1s[tid] * rsqrtf(bn1v[tid] + 1e-5f);
    inv1[tid] = iv;
    add1[tid] = (c1b[tid] - bn1m[tid]) * iv + bn1b[tid];
    h1[tid] = (half_t)0.f;                            // guard row t=-1
  } else if (tid < 64) {
    h1[321 * 32 + (tid - 32)] = (half_t)0.f;          // guard row t=320
  }
  if (tid < 16) {
    float iv = bn2s[tid] * rsqrtf(bn2v[tid] + 1e-5f);
    inv2[tid] = iv;
    add2[tid] = (c2b[tid] - bn2m[tid]) * iv + bn2b[tid];
    gbia[tid] = gb[tid];
  }
  __syncthreads();

  // hoist conv1 A-operands (weights): [tap][kstep][mtile]
  v16h a1[20];
#pragma unroll
  for (int k = 0; k < 5; ++k)
#pragma unroll
    for (int s = 0; s < 2; ++s)
#pragma unroll
      for (int m = 0; m < 2; ++m)
        a1[(k * 2 + s) * 2 + m] =
            load_a(w1l + (k * 32 + m * 16 + n16) * 64 + s * 32 + hi * 16);

  // ---- phase A: conv1 over T in 3 chunks of <=128 positions ----
  for (int cc = 0; cc < 3; ++cc) {
    const int c0 = cc * 128;
    const int ntiles = (cc == 2) ? 4 : 8;
    __syncthreads();
    // stage x_eeg rows c0-2 .. c0+129 (zero outside [0,T)) as f16
    for (int i4 = tid; i4 < (132 * 64) / 4; i4 += 256) {
      int idx = i4 * 4, row = idx >> 6, col = idx & 63;
      int t = c0 - 2 + row;
      float4 v = make_float4(0.f, 0.f, 0.f, 0.f);
      if (t >= 0 && t < TLEN)
        v = *(const float4*)(x_eeg + ((size_t)b * TLEN + t) * 64 + col);
      union { half_t h[4]; uint2 u; } pk;
      pk.h[0] = (half_t)v.x; pk.h[1] = (half_t)v.y;
      pk.h[2] = (half_t)v.z; pk.h[3] = (half_t)v.w;
      *(uint2*)(xec + idx) = pk.u;
    }
    __syncthreads();
    if (wid < ntiles) {
      const int t0 = c0 + wid * 16;
      v8f acc0 = {}; v8f acc1 = {};
#pragma unroll
      for (int k = 0; k < 5; ++k)
#pragma unroll
        for (int s = 0; s < 2; ++s) {
          // B(32x16): lane holds N=n16, lo lanes K(ch) 0..15, hi lanes 16..31
          v16h bm = *(const v16h*)(xec + (wid * 16 + n16 + k) * 64 + s * 32 + hi * 16);
          acc0 = wmma16(a1[(k * 2 + s) * 2 + 0], bm, acc0);
          acc1 = wmma16(a1[(k * 2 + s) * 2 + 1], bm, acc1);
        }
      const int trow = t0 + n16 + 1;           // +1 guard offset
      {
        union { half_t h[8]; uint4 u; } pk;
#pragma unroll
        for (int r = 0; r < 8; ++r) {
          int ch = hi * 8 + r;
          pk.h[r] = (half_t)fmaxf(acc0[r] * inv1[ch] + add1[ch], 0.f);
        }
        *(uint4*)(h1 + trow * 32 + hi * 8) = pk.u;
      }
      {
        union { half_t h[8]; uint4 u; } pk;
#pragma unroll
        for (int r = 0; r < 8; ++r) {
          int ch = 16 + hi * 8 + r;
          pk.h[r] = (half_t)fmaxf(acc1[r] * inv1[ch] + add1[ch], 0.f);
        }
        *(uint4*)(h1 + trow * 32 + 16 + hi * 8) = pk.u;
      }
    }
  }
  __syncthreads();

  // ---- phase B staging: conv2 weights, geo weights, x_geo (pad 18->32) ----
  for (int idx = tid; idx < 3 * 16 * 32; idx += 256) {
    int k = idx >> 9, o = (idx >> 5) & 15, i = idx & 31;
    w2l[idx] = (half_t)c2w[(o * 32 + i) * 3 + k];
  }
  for (int idx = tid; idx < 16 * 32; idx += 256) {
    int o = idx >> 5, i = idx & 31;
    wgl[idx] = (half_t)((i < 18) ? gw[o * 18 + i] : 0.f);
  }
  for (int idx = tid; idx < TLEN * 32; idx += 256) {
    int t = idx >> 5, i = idx & 31;
    xg[idx] = (half_t)((i < 18) ? x_geo[((size_t)b * TLEN + t) * 18 + i] : 0.f);
  }
  __syncthreads();

  v16h a2r[3];
#pragma unroll
  for (int k = 0; k < 3; ++k)
    a2r[k] = load_a(w2l + (k * 16 + n16) * 32 + hi * 16);
  v16h agr = load_a(wgl + n16 * 32 + hi * 16);

  // ---- phase B: conv2 + BN2 + ReLU and geo sigmoid, interleaved store ----
  for (int tt = wid; tt < 20; tt += 8) {
    const int t0 = tt * 16;
    v8f c2 = {};
#pragma unroll
    for (int k = 0; k < 3; ++k) {
      v16h bm = *(const v16h*)(h1 + (t0 + n16 + k) * 32 + hi * 16);
      c2 = wmma16(a2r[k], bm, c2);
    }
    v8f g = {};
    {
      v16h bg = *(const v16h*)(xg + (t0 + n16) * 32 + hi * 16);
      g = wmma16(agr, bg, g);
    }
    // enc row = 32 halfs: [2c]=eeg_encoded(c), [2c+1]=geo_mod(c)
    const size_t base = ((size_t)b * TLEN + (t0 + n16)) * 32 + hi * 16;
    union { half_t h[16]; uint4 u[2]; } pk;
#pragma unroll
    for (int r = 0; r < 8; ++r) {
      int ch = hi * 8 + r;
      pk.h[2 * r]     = (half_t)fmaxf(c2[r] * inv2[ch] + add2[ch], 0.f);
      float s = g[r] + gbia[ch];
      pk.h[2 * r + 1] = (half_t)(1.f / (1.f + __expf(-s)));
    }
    *(uint4*)(enc_out + base)     = pk.u[0];
    *(uint4*)(enc_out + base + 8) = pk.u[1];
  }
}

// ---------------------------------------------------------------------------
// Kernel 2: ALIF scan + windowed differential pooling.
// Thread = one (b,h) chain; block = 16 batches x 16 h.  Double-buffered
// GLOBAL_LOAD_ASYNC_TO_LDS_B128 (ASYNCcnt) fills chunk c+1 while chunk c is
// processed: [s_wait_asynccnt 0][barrier][issue next][compute] per chunk.
// ---------------------------------------------------------------------------
#define CHUNK_HALFS (16 * 16 * 32)          // 16 batches x 16 t x 32 halfs = 16KB

static __device__ __forceinline__ void issue_chunk(const half_t* enc, int b0,
                                                   int tid, int chunk, int buf) {
  // 1024 x b128 per chunk; 4 per thread.  GVS addressing: saddr(base)+vaddr.
  for (int i4 = tid; i4 < 1024; i4 += 256) {
    int hb = i4 * 8;                        // half index in chunk
    int lbi = hb >> 9, rem = hb & 511;      // 512 halfs per batch (16t x 32)
    int tt = rem >> 5, cc2 = rem & 31;
    unsigned goff = (unsigned)(((((b0 + lbi) * TLEN) + chunk * 16 + tt) * 32 + cc2) *
                               (int)sizeof(half_t));
    unsigned loff = (unsigned)(buf * (CHUNK_HALFS * 2) + hb * (int)sizeof(half_t));
    asm volatile("global_load_async_to_lds_b128 %0, %1, %2"
                 :: "v"(loff), "v"(goff), "s"(enc) : "memory");
  }
}

__global__ __launch_bounds__(256) void k_scan(
    const half_t* __restrict__ enc, const float* __restrict__ gamma,
    float* __restrict__ dp) {
  __shared__ __align__(16) half_t sb[2 * CHUNK_HALFS];   // 32KB double buffer
  const int tid = threadIdx.x;
  const int bi  = tid >> 4;
  const int h   = tid & 15;
  const int b0  = blockIdx.x * 16;
  const float gam = gamma[h];

  issue_chunk(enc, b0, tid, 0, 0);

  float mem = 0.f, eta = 0.f, mli = 0.f, spk = 0.f, a0 = 0.f;
  for (int ck = 0; ck < 20; ++ck) {                      // 16 steps per chunk
    const int buf = ck & 1;
    asm volatile("s_wait_asynccnt 0x0" ::: "memory");    // my fills landed
    __syncthreads();                                     // everyone's landed
    if (ck + 1 < 20) issue_chunk(enc, b0, tid, ck + 1, buf ^ 1);

    const half_t* base = sb + buf * CHUNK_HALFS + bi * 512 + h * 2;
    float acc = 0.f;
#pragma unroll 4
    for (int j = 0; j < 16; ++j) {
      union { unsigned u; half_t hh[2]; } pr;
      pr.u = *(const unsigned*)(base + j * 32);          // (eeg, geo) pair
      float e = (float)pr.hh[0], gm = (float)pr.hh[1];
      eta = 0.36f * eta + 0.64f * spk;
      float th = 0.5f + 1.8f * eta - gam * gm;
      mem = 0.8f * mem + e;
      spk = (mem - th >= 0.f) ? 1.f : 0.f;
      mem *= (1.f - spk);
      mli = 0.9f * mli + spk;
      acc += mli;
    }
    if ((ck & 1) == 0) {
      a0 = acc;                                          // first half-window
    } else {
      dp[(size_t)(b0 + bi) * 160 + h * 10 + (ck >> 1)] = (acc - a0) * 0.0625f;
    }
  }
}

// ---------------------------------------------------------------------------
// Kernel 3: fc1 (160->64, ReLU) + fc2 (64->4).  One wave per 16-batch tile,
// 8 waves/block.  fc1: 4 Mtiles x 5 Ksteps WMMA; LDS transpose; fc2: 2 WMMA.
// ---------------------------------------------------------------------------
__global__ __launch_bounds__(256) void k_fc(
    const float* __restrict__ dp, const float* __restrict__ w1,
    const float* __restrict__ b1, const float* __restrict__ w2,
    const float* __restrict__ b2, float* __restrict__ out) {
  __shared__ __align__(32) half_t fcbuf[8 * 16 * 64];  // [wave][batch][ch]
  const int tid = threadIdx.x;
  const int lane = tid & 31, wid = tid >> 5;
  const int n16 = lane & 15, hi = lane >> 4;
  const int batch0 = (blockIdx.x * 8 + wid) * 16;

  // B operands: lane N=n16, K = ks*32 + hi*16 + 0..15 (contiguous dp row)
  v16h bop[5];
  const float* drow = dp + (size_t)(batch0 + n16) * 160;
#pragma unroll
  for (int ks = 0; ks < 5; ++ks) {
    const float* p = drow + ks * 32 + hi * 16;
    v16h t;
#pragma unroll
    for (int j = 0; j < 16; ++j) t[j] = (half_t)p[j];
    bop[ks] = t;
  }

  v8f acc[4] = {};
#pragma unroll
  for (int m = 0; m < 4; ++m)
#pragma unroll
    for (int ks = 0; ks < 5; ++ks) {
      const float* p = w1 + (size_t)(m * 16 + n16) * 160 + ks * 32 + hi * 16;
      v16h a;
#pragma unroll
      for (int j = 0; j < 8; ++j) { a[j] = (half_t)p[j]; a[8 + j] = (half_t)p[8 + j]; }
      acc[m] = wmma16(a, bop[ks], acc[m]);
    }

  half_t* myrow = fcbuf + (wid * 16 + n16) * 64;
#pragma unroll
  for (int m = 0; m < 4; ++m) {
    union { half_t h[8]; uint4 u; } pk;
#pragma unroll
    for (int r = 0; r < 8; ++r) {
      int ch = m * 16 + hi * 8 + r;
      pk.h[r] = (half_t)fmaxf(acc[m][r] + b1[ch], 0.f);
    }
    *(uint4*)(myrow + m * 16 + hi * 8) = pk.u;
  }
  __syncthreads();

  v8f o = {};
#pragma unroll
  for (int ks = 0; ks < 2; ++ks) {
    v16h bm = *(const v16h*)(fcbuf + (wid * 16 + n16) * 64 + ks * 32 + hi * 16);
    v16h a;
    const int M = n16;  // rows 4..15 of A are zero padding
#pragma unroll
    for (int j = 0; j < 8; ++j) {
      int k0 = ks * 32 + hi * 16 + j;
      a[j]     = (M < 4) ? (half_t)w2[M * 64 + k0]      : (half_t)0.f;
      a[8 + j] = (M < 4) ? (half_t)w2[M * 64 + k0 + 8] : (half_t)0.f;
    }
    o = wmma16(a, bm, o);
  }
  if (hi == 0) {   // lanes 0-15 hold M=0..7 in VGPRs 0..7; M<4 valid
    float4 r;
    r.x = o[0] + b2[0]; r.y = o[1] + b2[1];
    r.z = o[2] + b2[2]; r.w = o[3] + b2[3];
    *(float4*)(out + (size_t)(batch0 + n16) * 4) = r;
  }
}

// ---------------------------------------------------------------------------
extern "C" void kernel_launch(void* const* d_in, const int* in_sizes, int n_in,
                              void* d_out, int out_size, void* d_ws, size_t ws_size,
                              hipStream_t stream) {
  const float* x_eeg = (const float*)d_in[0];
  const float* x_geo = (const float*)d_in[1];
  const float* c1w   = (const float*)d_in[2];
  const float* c1b   = (const float*)d_in[3];
  const float* bn1s  = (const float*)d_in[4];
  const float* bn1b  = (const float*)d_in[5];
  const float* bn1m  = (const float*)d_in[6];
  const float* bn1v  = (const float*)d_in[7];
  const float* c2w   = (const float*)d_in[8];
  const float* c2b   = (const float*)d_in[9];
  const float* bn2s  = (const float*)d_in[10];
  const float* bn2b  = (const float*)d_in[11];
  const float* bn2m  = (const float*)d_in[12];
  const float* bn2v  = (const float*)d_in[13];
  const float* gw    = (const float*)d_in[14];
  const float* gb    = (const float*)d_in[15];
  const float* gamma = (const float*)d_in[16];
  const float* fc1w  = (const float*)d_in[17];
  const float* fc1b  = (const float*)d_in[18];
  const float* fc2w  = (const float*)d_in[19];
  const float* fc2b  = (const float*)d_in[20];
  float* out = (float*)d_out;

  // workspace carve-up (needs >= 86,507,520 bytes)
  const size_t enc_bytes = (size_t)NB * TLEN * 32 * sizeof(half_t);  // 83,886,080
  half_t* enc_ws = (half_t*)d_ws;
  float*  dp_ws  = (float*)((char*)d_ws + enc_bytes);

  k_encode<<<NB, 256, 0, stream>>>(x_eeg, x_geo, c1w, c1b, bn1s, bn1b, bn1m, bn1v,
                                   c2w, c2b, bn2s, bn2b, bn2m, bn2v, gw, gb,
                                   enc_ws);
  k_scan<<<NB / 16, 256, 0, stream>>>(enc_ws, gamma, dp_ws);
  k_fc<<<NB / 128, 256, 0, stream>>>(dp_ws, fc1w, fc1b, fc2w, fc2b, out);
}
